// LoFTREncoderLayer_88373247082501
// MI455X (gfx1250) — matrix operand order repaired and checked by reading, not compiled, not run on hardware
//
#include <hip/hip_runtime.h>
#include <hip/hip_bf16.h>
#include <math.h>

typedef __bf16 bf16_t;
typedef __attribute__((ext_vector_type(16))) __bf16 v16bf;
typedef __attribute__((ext_vector_type(8)))  float  v8f;

#define NHEAD 8
#define HDIM  32
#define LSEQ  4096
#define CDIM  256
#define BATCH 8

// ---------------------------------------------------------------------------
// Weight f32 [K][N] -> bf16 [N][K] transpose-convert (32x32 tiles via LDS).
// Lets GEMM stage B tiles with contiguous b128 loads/stores (no ds scatter).
// ---------------------------------------------------------------------------
__global__ __launch_bounds__(256)
void wtconv_kernel(const float* __restrict__ w, bf16_t* __restrict__ o, int K, int N) {
    __shared__ float t[32][33];
    const int tx = threadIdx.x, ty = threadIdx.y;
    const int k0 = blockIdx.x * 32, n0 = blockIdx.y * 32;
    #pragma unroll
    for (int i = ty; i < 32; i += 8)
        t[i][tx] = w[(size_t)(k0 + i) * N + n0 + tx];
    __syncthreads();
    #pragma unroll
    for (int i = ty; i < 32; i += 8)
        o[(size_t)(n0 + i) * K + k0 + tx] = (bf16_t)t[tx][i];
}

// ---------------------------------------------------------------------------
// [B, C, L] f32 -> [B, L, C] bf16 transpose-pack (32x32 tiles via LDS)
// ---------------------------------------------------------------------------
__global__ __launch_bounds__(256)
void transpose_pack_kernel(const float* __restrict__ src, bf16_t* __restrict__ dst, int ldo) {
    __shared__ float tile[32][33];
    const int tx = threadIdx.x, ty = threadIdx.y;
    const int l0 = blockIdx.x * 32, c0 = blockIdx.y * 32, b = blockIdx.z;
    const float* s = src + ((size_t)b * CDIM) * LSEQ;
    #pragma unroll
    for (int i = ty; i < 32; i += 8)
        tile[i][tx] = s[(size_t)(c0 + i) * LSEQ + l0 + tx];
    __syncthreads();
    bf16_t* d = dst + ((size_t)b * LSEQ) * ldo;
    #pragma unroll
    for (int i = ty; i < 32; i += 8)
        d[(size_t)(l0 + i) * ldo + c0 + tx] = (bf16_t)tile[tx][i];
}

// ---------------------------------------------------------------------------
// Tiled WMMA GEMM: D[M,N] = A[M,K](bf16, lda) @ Bt[N,K](bf16, pre-transposed)
// BM=BN=128, BK=32; 256 threads = 8 waves, wave tile 32x64.
// Register-staged double-buffered pipeline: next tile's global_load_b128s are
// issued before the single per-iteration barrier so HBM/L2 latency overlaps
// the 8 WMMAs; global_prefetch_b8 runs two K-steps ahead.
// EPI: 0 = f32 out, 1 = elu+1 -> bf16, 2 = *scale -> bf16, 3 = relu -> bf16
// ---------------------------------------------------------------------------
template<int EPI>
__global__ __launch_bounds__(256)
void gemm_wmma_kernel(const bf16_t* __restrict__ A, int lda,
                      const bf16_t* __restrict__ Bt,
                      void* __restrict__ D, int ldd,
                      int K, float scale) {
    __shared__ bf16_t As[2][128][48];   // row length 96B (16B-aligned)
    __shared__ bf16_t Bs[2][128][48];   // [n][k]

    const int tid  = threadIdx.x;
    const int lane = tid & 31, wid = tid >> 5;
    const int wm = wid & 3, wn = wid >> 2;       // 4 x 2 wave grid
    const size_t m0 = (size_t)blockIdx.y * 128;
    const int    n0 = blockIdx.x * 128;

    v8f acc[2][4] = {};

    const int srow  = tid >> 1;                  // 128 rows
    const int shalf = (tid & 1) * 16;            // two 16-element halves
    const int mr = lane & 15;
    const int kq = lane >> 4;                    // 0 or 1 (lane half)

    const bf16_t* aptr = A + (m0 + srow) * lda + shalf;
    const bf16_t* bptr = Bt + (size_t)(n0 + srow) * K + shalf;

    uint4 ra = *(const uint4*)aptr;
    uint4 rb = *(const uint4*)bptr;

    int buf = 0;
    for (int k0 = 0; k0 < K; k0 += 32) {
        *(uint4*)&As[buf][srow][shalf] = ra;
        *(uint4*)&Bs[buf][srow][shalf] = rb;
        if (k0 + 32 < K) {                        // stage next tile into regs
            ra = *(const uint4*)(aptr + k0 + 32);
            rb = *(const uint4*)(bptr + k0 + 32);
        }
        if (k0 + 64 < K) {                        // L2/L0 prefetch 2 steps out
            __builtin_prefetch(aptr + k0 + 64, 0, 3);
            __builtin_prefetch(bptr + k0 + 64, 0, 3);
        }
        __syncthreads();

        union Frag { uint4 u[2]; v16bf v; };
        Frag af[2], bfr[4];
        #pragma unroll
        for (int mt = 0; mt < 2; ++mt) {
            // A layout: VGPR0-3 = K[kq*8 .. +7], VGPR4-7 = K[kq*8+16 .. +23]
            const uint4* p = (const uint4*)&As[buf][wm * 32 + mt * 16 + mr][0];
            af[mt].u[0] = p[kq];
            af[mt].u[1] = p[kq + 2];
        }
        #pragma unroll
        for (int nt = 0; nt < 4; ++nt) {
            // B layout: VGPR0-3 = K[kq*8 .. +7], VGPR4-7 = K[kq*8+16 .. +23]
            const uint4* p = (const uint4*)&Bs[buf][wn * 64 + nt * 16 + mr][0];
            bfr[nt].u[0] = p[kq];
            bfr[nt].u[1] = p[kq + 2];
        }
        #pragma unroll
        for (int mt = 0; mt < 2; ++mt)
            #pragma unroll
            for (int nt = 0; nt < 4; ++nt)
                acc[mt][nt] = __builtin_amdgcn_wmma_f32_16x16x32_bf16(
                    false, af[mt].v, false, bfr[nt].v, (short)0, acc[mt][nt],
                    false, false);
        buf ^= 1;
    }

    // epilogue: C/D layout — VGPR r: row = r + 8*(lane>=16), col = lane&15
    const int hi = kq * 8;
    #pragma unroll
    for (int mt = 0; mt < 2; ++mt)
        #pragma unroll
        for (int nt = 0; nt < 4; ++nt)
            #pragma unroll
            for (int r = 0; r < 8; ++r) {
                size_t gm = m0 + wm * 32 + mt * 16 + r + hi;
                int    gn = n0 + wn * 64 + nt * 16 + mr;
                float  v  = acc[mt][nt][r];
                if constexpr (EPI == 0) {
                    ((float*)D)[gm * ldd + gn] = v;
                } else {
                    float o;
                    if constexpr (EPI == 1)      o = (v > 0.0f) ? v + 1.0f : __expf(v);
                    else if constexpr (EPI == 2) o = v * scale;
                    else                         o = (v > 0.0f) ? v : 0.0f;
                    ((bf16_t*)D)[gm * ldd + gn] = (bf16_t)o;
                }
            }
}

// ---------------------------------------------------------------------------
// KV[b,h][32][32] = sum_l K[b,l,h,:]^T outer V[b,l,h,:]   (WMMA over L)
// grid = B*NHEAD, 256 threads = 8 waves, each wave covers 16 L-chunks of 32.
// ---------------------------------------------------------------------------
__global__ __launch_bounds__(256)
void kv_kernel(const bf16_t* __restrict__ Kb, const bf16_t* __restrict__ Vb,
               float* __restrict__ KV) {
    __shared__ bf16_t Kt[8][32][48];   // per-wave [d][l] transposed chunk
    __shared__ bf16_t Vt[8][32][48];
    __shared__ float  red[32][32];

    const int tid = threadIdx.x, lane = tid & 31, wid = tid >> 5;
    const int b = blockIdx.x >> 3, h = blockIdx.x & 7;
    const size_t base = ((size_t)b * LSEQ) * CDIM + h * HDIM;

    for (int i = tid; i < 1024; i += 256) (&red[0][0])[i] = 0.0f;

    v8f acc[2][2] = {};
    const int mr = lane & 15, kq = lane >> 4;

    for (int c = wid; c < LSEQ / 32; c += 8) {
        __syncthreads();
        // stage chunk transposed: lane owns row l = c*32 + lane
        {
            const bf16_t* kp = Kb + base + (size_t)(c * 32 + lane) * CDIM;
            const bf16_t* vp = Vb + base + (size_t)(c * 32 + lane) * CDIM;
            union { uint4 u; bf16_t h[16]; } tk[2], tv[2];
            tk[0].u = ((const uint4*)kp)[0]; tk[1].u = ((const uint4*)kp)[1];
            tv[0].u = ((const uint4*)vp)[0]; tv[1].u = ((const uint4*)vp)[1];
            #pragma unroll
            for (int d = 0; d < 32; ++d) {
                Kt[wid][d][lane] = tk[d >> 4].h[d & 15];
                Vt[wid][d][lane] = tv[d >> 4].h[d & 15];
            }
        }
        __syncthreads();

        union Frag { uint4 u[2]; v16bf v; };
        Frag af[2], bfr[2];
        #pragma unroll
        for (int mt = 0; mt < 2; ++mt) {
            const uint4* p = (const uint4*)&Kt[wid][mt * 16 + mr][0];
            af[mt].u[0] = p[kq];
            af[mt].u[1] = p[kq + 2];
        }
        #pragma unroll
        for (int nt = 0; nt < 2; ++nt) {
            const uint4* p = (const uint4*)&Vt[wid][nt * 16 + mr][0];
            bfr[nt].u[0] = p[kq];
            bfr[nt].u[1] = p[kq + 2];
        }
        #pragma unroll
        for (int mt = 0; mt < 2; ++mt)
            #pragma unroll
            for (int nt = 0; nt < 2; ++nt)
                acc[mt][nt] = __builtin_amdgcn_wmma_f32_16x16x32_bf16(
                    false, af[mt].v, false, bfr[nt].v, (short)0, acc[mt][nt],
                    false, false);
    }
    __syncthreads();

    const int hi = kq * 8;
    #pragma unroll
    for (int mt = 0; mt < 2; ++mt)
        #pragma unroll
        for (int nt = 0; nt < 2; ++nt)
            #pragma unroll
            for (int r = 0; r < 8; ++r)
                atomicAdd(&red[mt * 16 + r + hi][nt * 16 + mr], acc[mt][nt][r]);
    __syncthreads();

    float* out = KV + (size_t)blockIdx.x * 1024;
    for (int i = tid; i < 1024; i += 256) out[i] = (&red[0][0])[i];
}

// ---------------------------------------------------------------------------
// Ksum[b,h,d] = sum_l K[b,l,h,d]
// ---------------------------------------------------------------------------
__global__ __launch_bounds__(256)
void ksum_kernel(const bf16_t* __restrict__ Kb, float* __restrict__ Ksum) {
    __shared__ float red[8][32];
    const int tid = threadIdx.x, d = tid & 31, g = tid >> 5;
    const int b = blockIdx.x >> 3, h = blockIdx.x & 7;
    const size_t base = ((size_t)b * LSEQ) * CDIM + h * HDIM + d;
    float acc = 0.0f;
    for (int l = g; l < LSEQ; l += 8) acc += (float)Kb[base + (size_t)l * CDIM];
    red[g][d] = acc;
    __syncthreads();
    if (tid < 32) {
        float s = 0.0f;
        #pragma unroll
        for (int i = 0; i < 8; ++i) s += red[i][tid];
        Ksum[blockIdx.x * 32 + tid] = s;
    }
}

// ---------------------------------------------------------------------------
// Zs[b,l,h] = L / (Q[b,l,h,:] . Ksum[b,h,:] + eps)     (L folded in here)
// ---------------------------------------------------------------------------
__global__ void z_kernel(const bf16_t* __restrict__ Qb, const float* __restrict__ Ksum,
                         float* __restrict__ Zs) {
    const int idx = blockIdx.x * blockDim.x + threadIdx.x;  // B*L*NHEAD
    const int h = idx & 7;
    const int bl = idx >> 3;
    const int b = bl >> 12;
    const bf16_t* q  = Qb + (size_t)bl * CDIM + h * HDIM;
    const float*  ks = Ksum + (b * NHEAD + h) * HDIM;
    float dot = 0.0f;
    #pragma unroll
    for (int d = 0; d < HDIM; ++d) dot += (float)q[d] * ks[d];
    Zs[idx] = (float)LSEQ / (dot + 1e-6f);
}

// ---------------------------------------------------------------------------
// msg_pre[b,l,h*32+:32] = (Q[b,l,h,:] @ KV[b,h]) * Zs[b,l,h]   -> bf16
// grid = (L/128, B*NHEAD); wave handles 16 rows; K=32 single WMMA step.
// ---------------------------------------------------------------------------
__global__ __launch_bounds__(256)
void msg_kernel(const bf16_t* __restrict__ Qb, const float* __restrict__ KV,
                const float* __restrict__ Zs, bf16_t* __restrict__ Mp) {
    const int lane = threadIdx.x & 31, wid = threadIdx.x >> 5;
    const int b = blockIdx.y >> 3, h = blockIdx.y & 7;
    const int mr = lane & 15, kq = lane >> 4;
    const int row0 = blockIdx.x * 128 + wid * 16;
    const size_t blrow = (size_t)b * LSEQ + row0;

    union Frag { uint4 u[2]; v16bf v; };
    Frag a;
    const uint4* ap = (const uint4*)(Qb + (blrow + mr) * CDIM + h * HDIM);
    a.u[0] = ap[kq];
    a.u[1] = ap[kq + 2];

    const float* kv = KV + (size_t)blockIdx.y * 1024;
    Frag bfr[2];
    #pragma unroll
    for (int nt = 0; nt < 2; ++nt) {
        union { uint4 u[2]; bf16_t hh[16]; } t;
        #pragma unroll
        for (int i = 0; i < 16; ++i)
            t.hh[i] = (bf16_t)kv[(kq * 8 + (i & 7) + (i >> 3) * 16) * 32 + nt * 16 + mr];
        bfr[nt].u[0] = t.u[0];
        bfr[nt].u[1] = t.u[1];
    }

    v8f acc[2] = {};
    #pragma unroll
    for (int nt = 0; nt < 2; ++nt)
        acc[nt] = __builtin_amdgcn_wmma_f32_16x16x32_bf16(
            false, a.v, false, bfr[nt].v, (short)0, acc[nt], false, false);

    const int hi = kq * 8;
    #pragma unroll
    for (int r = 0; r < 8; ++r) {
        size_t gm = blrow + r + hi;
        float  zs = Zs[gm * NHEAD + h];
        #pragma unroll
        for (int nt = 0; nt < 2; ++nt)
            Mp[gm * CDIM + h * HDIM + nt * 16 + mr] = (bf16_t)(acc[nt][r] * zs);
    }
}

// ---------------------------------------------------------------------------
// LayerNorm helpers (one wave32 per 256-wide row)
// ---------------------------------------------------------------------------
__device__ inline float wave_sum(float v) {
    #pragma unroll
    for (int m = 16; m > 0; m >>= 1) v += __shfl_xor(v, m, 32);
    return v;
}

// LN1: ln(msgW)*g1+b1 -> bf16 into cat[:, 256:512]
__global__ __launch_bounds__(256)
void ln1_kernel(const float* __restrict__ X, const float* __restrict__ g,
                const float* __restrict__ bb, bf16_t* __restrict__ cat) {
    const int lane = threadIdx.x & 31, wid = threadIdx.x >> 5;
    const size_t row = (size_t)blockIdx.x * 8 + wid;
    const float* xr = X + row * CDIM;
    float v[8], s = 0.0f;
    #pragma unroll
    for (int i = 0; i < 8; ++i) { v[i] = xr[lane * 8 + i]; s += v[i]; }
    const float mean = wave_sum(s) * (1.0f / 256.0f);
    float q = 0.0f;
    #pragma unroll
    for (int i = 0; i < 8; ++i) { float d = v[i] - mean; q += d * d; }
    const float rstd = rsqrtf(wave_sum(q) * (1.0f / 256.0f) + 1e-5f);
    bf16_t* o = cat + row * 512 + 256;
    #pragma unroll
    for (int i = 0; i < 8; ++i) {
        int c = lane * 8 + i;
        o[c] = (bf16_t)((v[i] - mean) * rstd * g[c] + bb[c]);
    }
}

// final: out[b,c,l] = x[b,c,l] + (ln(y2)*g2+b2)[c]
__global__ __launch_bounds__(256)
void final_kernel(const float* __restrict__ Y, const float* __restrict__ g,
                  const float* __restrict__ bb, const float* __restrict__ x,
                  float* __restrict__ out) {
    const int lane = threadIdx.x & 31, wid = threadIdx.x >> 5;
    const size_t row = (size_t)blockIdx.x * 8 + wid;   // b*L + l
    const float* yr = Y + row * CDIM;
    float v[8], s = 0.0f;
    #pragma unroll
    for (int i = 0; i < 8; ++i) { v[i] = yr[lane * 8 + i]; s += v[i]; }
    const float mean = wave_sum(s) * (1.0f / 256.0f);
    float q = 0.0f;
    #pragma unroll
    for (int i = 0; i < 8; ++i) { float d = v[i] - mean; q += d * d; }
    const float rstd = rsqrtf(wave_sum(q) * (1.0f / 256.0f) + 1e-5f);
    const int b = (int)(row >> 12), l = (int)(row & 4095);
    #pragma unroll
    for (int i = 0; i < 8; ++i) {
        int c = lane * 8 + i;
        size_t gi = ((size_t)(b * CDIM + c)) * LSEQ + l;
        out[gi] = x[gi] + (v[i] - mean) * rstd * g[c] + bb[c];
    }
}

// ---------------------------------------------------------------------------
// Host launcher
// ---------------------------------------------------------------------------
extern "C" void kernel_launch(void* const* d_in, const int* in_sizes, int n_in,
                              void* d_out, int out_size, void* d_ws, size_t ws_size,
                              hipStream_t stream) {
    const float* x      = (const float*)d_in[0];
    const float* source = (const float*)d_in[1];
    const float* Wq = (const float*)d_in[2];
    const float* Wk = (const float*)d_in[3];
    const float* Wv = (const float*)d_in[4];
    const float* Wm = (const float*)d_in[5];
    const float* W1 = (const float*)d_in[6];
    const float* W2 = (const float*)d_in[7];
    const float* g1 = (const float*)d_in[8];
    const float* b1 = (const float*)d_in[9];
    const float* g2 = (const float*)d_in[10];
    const float* b2 = (const float*)d_in[11];

    const size_t Mtot = (size_t)BATCH * LSEQ;  // 32768
    char* p = (char*)d_ws;
    auto take = [&](size_t bytes) -> char* {
        char* r = p;
        p += (bytes + 255) & ~(size_t)255;
        return r;
    };

    bf16_t* ssb   = (bf16_t*)take(Mtot * CDIM * 2);       // source^T bf16
    bf16_t* cat   = (bf16_t*)take(Mtot * 512 * 2);        // [xs | ln1] bf16
    bf16_t* Qb    = (bf16_t*)take(Mtot * CDIM * 2);
    bf16_t* Kb    = (bf16_t*)take(Mtot * CDIM * 2);
    bf16_t* Vb    = (bf16_t*)take(Mtot * CDIM * 2);
    bf16_t* Mp    = (bf16_t*)take(Mtot * CDIM * 2);       // msg_pre bf16
    bf16_t* y1b   = (bf16_t*)take(Mtot * 512 * 2);        // relu(cat@W1)
    float*  bigf  = (float*)take(Mtot * CDIM * 4);        // msgW then y2
    bf16_t* Wqt   = (bf16_t*)take(CDIM * CDIM * 2);       // weights, [N][K] bf16
    bf16_t* Wkt   = (bf16_t*)take(CDIM * CDIM * 2);
    bf16_t* Wvt   = (bf16_t*)take(CDIM * CDIM * 2);
    bf16_t* Wmt   = (bf16_t*)take(CDIM * CDIM * 2);
    bf16_t* W1t   = (bf16_t*)take(512 * 512 * 2);
    bf16_t* W2t   = (bf16_t*)take(512 * CDIM * 2);
    float*  KV    = (float*)take(BATCH * NHEAD * HDIM * HDIM * 4);
    float*  Ksum  = (float*)take(BATCH * NHEAD * HDIM * 4);
    float*  Zs    = (float*)take(Mtot * NHEAD * 4);

    dim3 tb(32, 8);

    // 1) weights -> transposed bf16 [N][K]
    wtconv_kernel<<<dim3(8, 8),   tb, 0, stream>>>(Wq, Wqt, CDIM, CDIM);
    wtconv_kernel<<<dim3(8, 8),   tb, 0, stream>>>(Wk, Wkt, CDIM, CDIM);
    wtconv_kernel<<<dim3(8, 8),   tb, 0, stream>>>(Wv, Wvt, CDIM, CDIM);
    wtconv_kernel<<<dim3(8, 8),   tb, 0, stream>>>(Wm, Wmt, CDIM, CDIM);
    wtconv_kernel<<<dim3(16, 16), tb, 0, stream>>>(W1, W1t, 512, 512);
    wtconv_kernel<<<dim3(16, 8),  tb, 0, stream>>>(W2, W2t, 512, CDIM);

    // 2) transpose-pack inputs
    transpose_pack_kernel<<<dim3(LSEQ / 32, CDIM / 32, BATCH), tb, 0, stream>>>(x, cat, 512);
    transpose_pack_kernel<<<dim3(LSEQ / 32, CDIM / 32, BATCH), tb, 0, stream>>>(source, ssb, CDIM);

    // 3) q/k/v projections (fused elu+1 for Q,K; fused 1/L for V)
    dim3 g2d(CDIM / 128, (unsigned)(Mtot / 128));
    gemm_wmma_kernel<1><<<g2d, 256, 0, stream>>>(cat, 512, Wqt, Qb, CDIM, CDIM, 0.0f);
    gemm_wmma_kernel<1><<<g2d, 256, 0, stream>>>(ssb, CDIM, Wkt, Kb, CDIM, CDIM, 0.0f);
    gemm_wmma_kernel<2><<<g2d, 256, 0, stream>>>(ssb, CDIM, Wvt, Vb, CDIM, CDIM,
                                                 1.0f / (float)LSEQ);

    // 4) attention core
    ksum_kernel<<<BATCH * NHEAD, 256, 0, stream>>>(Kb, Ksum);
    kv_kernel<<<BATCH * NHEAD, 256, 0, stream>>>(Kb, Vb, KV);
    z_kernel<<<(unsigned)(Mtot * NHEAD / 256), 256, 0, stream>>>(Qb, Ksum, Zs);
    msg_kernel<<<dim3(LSEQ / 128, BATCH * NHEAD), 256, 0, stream>>>(Qb, KV, Zs, Mp);

    // 5) merge projection + LN1 (writes into cat second half)
    gemm_wmma_kernel<0><<<g2d, 256, 0, stream>>>(Mp, CDIM, Wmt, bigf, CDIM, CDIM, 0.0f);
    ln1_kernel<<<(unsigned)(Mtot / 8), 256, 0, stream>>>(bigf, g1, b1, cat);

    // 6) FFN
    dim3 gW1(512 / 128, (unsigned)(Mtot / 128));
    gemm_wmma_kernel<3><<<gW1, 256, 0, stream>>>(cat, 512, W1t, y1b, 512, 512, 0.0f);
    gemm_wmma_kernel<0><<<g2d, 256, 0, stream>>>(y1b, 512, W2t, bigf, CDIM, 512, 0.0f);

    // 7) LN2 + residual + transpose back to [B, C, H, W]
    final_kernel<<<(unsigned)(Mtot / 8), 256, 0, stream>>>(bigf, g2, b2, x, (float*)d_out);
}
